// SparseChannelLinear_77438260346927
// MI455X (gfx1250) — matrix operand order repaired and checked by reading, not compiled, not run on hardware
//
#include <hip/hip_runtime.h>

// ---------------------------------------------------------------------------
// SparseChannelLinear on MI455X (gfx1250, wave32, WMMA + async-to-LDS)
//
// y[t, active_rows[r]] = sum_c x[t, active_cols[c]] * W[r,c]*ALPHA + bias
// y[t, other]          = bias[other]
//
// M=8192, N=2048, K=2048 -> 68.7 GFLOP; HBM traffic ~400 MB -> ~17us at
// 23.3 TB/s, so we need the bf16 WMMA rate (v_wmma_f32_16x16x32_bf16).
// bf16 Xc (33.5 MB) + bf16 W (8.4 MB) live in the 192 MB L2 across tiles.
// Tile staging uses GLOBAL_LOAD_ASYNC_TO_LDS_B128 (ASYNCcnt) so no VGPR
// round-trip: global -> LDS directly, drained with s_wait_asynccnt.
// ---------------------------------------------------------------------------

typedef __bf16 bf16_t;
typedef __attribute__((ext_vector_type(16))) __bf16 v16bf;
typedef __attribute__((ext_vector_type(8)))  __bf16 v8bf;
typedef __attribute__((ext_vector_type(4)))  __bf16 v4bf;
typedef __attribute__((ext_vector_type(8)))  float  v8f;

#define ALPHA_F   0.05f
#define FULL_IN   4096
#define FULL_OUT  4096
#define NROWS     2048      // active output channels (N)
#define NCOLS     2048      // active input channels  (K)
#define NTOK      8192      // B*S tokens             (M)

// Workgroup-relative LDS byte address of a __shared__ object (what the
// VDST operand of global_load_async_to_lds_* expects).
__device__ __forceinline__ unsigned lds_addr(const void* p) {
  return (unsigned)(unsigned long long)
      (const __attribute__((address_space(3))) char*)p;
}

__device__ __forceinline__ void wait_asynccnt0() {
#if __has_builtin(__builtin_amdgcn_s_wait_asynccnt)
  __builtin_amdgcn_s_wait_asynccnt(0);
#else
  asm volatile("s_wait_asynccnt 0x0" ::: "memory");
#endif
}

// ---------------------------------------------------------------------------
// Kernel 1: gather live input channels and convert to bf16.
// active_cols is sorted -> consecutive c hit nearby addresses in x's 16KB row.
// ---------------------------------------------------------------------------
__global__ __launch_bounds__(256) void k_gather_x(
    const float* __restrict__ x, const int* __restrict__ cols,
    bf16_t* __restrict__ xc)
{
  const int i = blockIdx.x * 256 + threadIdx.x;       // over NTOK*NCOLS
  const int t = i >> 11;                              // / NCOLS
  const int c = i & (NCOLS - 1);
  xc[i] = (bf16_t)x[(size_t)t * FULL_IN + cols[c]];
}

// ---------------------------------------------------------------------------
// Kernel 2: fold ALPHA into W and convert to bf16 (row-major [N][K]).
// ---------------------------------------------------------------------------
__global__ __launch_bounds__(256) void k_conv_w(
    const float* __restrict__ w, bf16_t* __restrict__ wb)
{
  const int i = blockIdx.x * 256 + threadIdx.x;       // over NROWS*NCOLS/4
  const float4 f = ((const float4*)w)[i];
  v4bf o;
  o[0] = (bf16_t)(f.x * ALPHA_F);
  o[1] = (bf16_t)(f.y * ALPHA_F);
  o[2] = (bf16_t)(f.z * ALPHA_F);
  o[3] = (bf16_t)(f.w * ALPHA_F);
  ((v4bf*)wb)[i] = o;
}

// ---------------------------------------------------------------------------
// Kernel 3: broadcast bias into the full output (inactive rows keep this).
// ---------------------------------------------------------------------------
__global__ __launch_bounds__(256) void k_fill_bias(
    const float* __restrict__ bias, float* __restrict__ out)
{
  const int i = blockIdx.x * 256 + threadIdx.x;       // over NTOK*FULL_OUT/4
  ((float4*)out)[i] = ((const float4*)bias)[i & (FULL_OUT / 4 - 1)];
}

// ---------------------------------------------------------------------------
// Kernel 4: tiled bf16 WMMA GEMM + scatter epilogue.
//   block tile: 128(M) x 128(N), K-step 32, 8 waves (wave32) = 256 threads
//   wave grid 4x2 -> each wave: 32(M) x 64(N) = 2x4 v_wmma accumulators
//   LDS double buffer, rows padded to 40 bf16 (80B) -> conflict-free b128
//   Staging: global_load_async_to_lds_b128 (GVS: SGPR base + VGPR byte off)
// ---------------------------------------------------------------------------
#define TK     32
#define LDPAD  40

__global__ __launch_bounds__(256, 1) void k_gemm(
    const bf16_t* __restrict__ xc,   // [NTOK][NCOLS] bf16
    const bf16_t* __restrict__ wb,   // [NROWS][NCOLS] bf16 (alpha folded)
    const int*    __restrict__ rows, // active_rows (sorted)
    const float*  __restrict__ bias,
    float*        __restrict__ out)  // [NTOK][FULL_OUT]
{
  __shared__ bf16_t sA[2][128][LDPAD];   // M x K tile
  __shared__ bf16_t sB[2][128][LDPAD];   // N x K tile
  const int nb    = blockIdx.x;          // 16 N-blocks
  const int mb    = blockIdx.y;          // 64 M-blocks
  const int tid   = threadIdx.x;
  const int lane  = tid & 31;
  const int wave  = tid >> 5;
  const int wm    = wave & 3;            // 0..3 -> 32-row strip
  const int wn    = wave >> 2;           // 0..1 -> 64-col strip
  const int mbase = mb * 128;
  const int nbase = nb * 128;
  const int lhalf = lane >> 4;           // 0 | 1 (lane half)
  const int l15   = lane & 15;

  // Async tile loader: 128 rows x 32 cols per matrix = 512 chunks of 16B,
  // 256 threads x 2 chunks each. global -> LDS direct (no VGPR staging).
  auto loadTileAsync = [&](int s, int k0) {
#pragma unroll
    for (int r = 0; r < 2; ++r) {
      const int ch  = tid + r * 256;
      const int row = ch >> 2;
      const int seg = (ch & 3) * 8;                       // 8 bf16 = 16B
      const unsigned aoff =
          (unsigned)(((mbase + row) * NCOLS + k0 + seg) * 2);
      const unsigned boff =
          (unsigned)(((nbase + row) * NCOLS + k0 + seg) * 2);
      const unsigned lA = lds_addr(&sA[s][row][seg]);
      const unsigned lB = lds_addr(&sB[s][row][seg]);
      asm volatile("global_load_async_to_lds_b128 %0, %1, %2"
                   :: "v"(lA), "v"(aoff), "s"(xc) : "memory");
      asm volatile("global_load_async_to_lds_b128 %0, %1, %2"
                   :: "v"(lB), "v"(boff), "s"(wb) : "memory");
    }
  };

  v8f acc[2][4];
#pragma unroll
  for (int mi = 0; mi < 2; ++mi)
#pragma unroll
    for (int ni = 0; ni < 4; ++ni)
      acc[mi][ni] = (v8f)0.0f;

  // Epilogue scatter targets: column index + bias per lane (depend on lane&15
  // only). active_rows is sorted -> each 16-lane group hits a narrow window.
  int   gcol[4];
  float gb[4];
#pragma unroll
  for (int ni = 0; ni < 4; ++ni) {
    const int gn = nbase + wn * 64 + ni * 16 + l15;
    gcol[ni] = rows[gn];
    gb[ni]   = bias[gcol[ni]];
  }

  loadTileAsync(0, 0);
  wait_asynccnt0();
  __syncthreads();

  const int NK = NCOLS / TK;             // 64 K-steps
  for (int kt = 0; kt < NK; ++kt) {
    const int s = kt & 1;
    if (kt + 1 < NK) loadTileAsync(s ^ 1, (kt + 1) * TK);  // fill other buffer

    // A fragment (16x32 bf16): lane half selects K {0..7,16..23}/{8..15,24..31}
    const int koffA = lhalf * 8;
    const int mrow  = wm * 32 + l15;
    v16bf a[2], b[4];
#pragma unroll
    for (int mi = 0; mi < 2; ++mi) {
      union { v16bf v; v8bf h[2]; } u;
      u.h[0] = *(const v8bf*)&sA[s][mrow + mi * 16][koffA];
      u.h[1] = *(const v8bf*)&sA[s][mrow + mi * 16][koffA + 16];
      a[mi] = u.v;
    }
    // B fragment (32x16 bf16): N = lane&15, lane half selects K {0..7,16..23}/{8..15,24..31}
    const int kb = lhalf * 8;
#pragma unroll
    for (int ni = 0; ni < 4; ++ni) {
      const int n = wn * 64 + ni * 16 + l15;
      union { v16bf v; v8bf h[2]; } u;
      u.h[0] = *(const v8bf*)&sB[s][n][kb];
      u.h[1] = *(const v8bf*)&sB[s][n][kb + 16];
      b[ni] = u.v;
    }

#pragma unroll
    for (int mi = 0; mi < 2; ++mi)
#pragma unroll
      for (int ni = 0; ni < 4; ++ni)
        acc[mi][ni] = __builtin_amdgcn_wmma_f32_16x16x32_bf16(
            /*neg_a=*/false, a[mi], /*neg_b=*/false, b[ni],
            /*c_mod=*/(short)0, acc[mi][ni],
            /*reuse_a=*/false, /*reuse_b=*/false);

    // Drain this iteration's async fills, then hand buffers over.
    wait_asynccnt0();
    __syncthreads();
  }

  // Epilogue: C/D layout (§7.12.2): row M = vgpr + 8*(lane>=16), col = lane&15.
#pragma unroll
  for (int mi = 0; mi < 2; ++mi) {
#pragma unroll
    for (int ni = 0; ni < 4; ++ni) {
#pragma unroll
      for (int v = 0; v < 8; ++v) {
        const int token = mbase + wm * 32 + mi * 16 + v + 8 * lhalf;
        out[(size_t)token * FULL_OUT + gcol[ni]] = acc[mi][ni][v] + gb[ni];
      }
    }
  }
}

// ---------------------------------------------------------------------------
// Host launcher
// ---------------------------------------------------------------------------
extern "C" void kernel_launch(void* const* d_in, const int* in_sizes, int n_in,
                              void* d_out, int out_size, void* d_ws, size_t ws_size,
                              hipStream_t stream)
{
  (void)in_sizes; (void)n_in; (void)out_size; (void)ws_size;

  const float* x    = (const float*)d_in[0];   // [4,2048,4096] f32
  const float* w    = (const float*)d_in[1];   // [2048,2048]  f32
  const float* bias = (const float*)d_in[2];   // [4096]       f32
  const int*   rows = (const int*)d_in[3];     // [2048] sorted
  const int*   cols = (const int*)d_in[4];     // [2048] sorted
  float*       out  = (float*)d_out;           // [4,2048,4096] f32

  bf16_t* xc = (bf16_t*)d_ws;                                    // 33.5 MB
  bf16_t* wb = (bf16_t*)((char*)d_ws + (size_t)NTOK * NCOLS * 2);//  8.4 MB

  k_gather_x<<<(NTOK * NCOLS) / 256, 256, 0, stream>>>(x, cols, xc);
  k_conv_w  <<<(NROWS * NCOLS / 4) / 256, 256, 0, stream>>>(w, wb);
  k_fill_bias<<<((size_t)NTOK * FULL_OUT / 4) / 256, 256, 0, stream>>>(bias, out);
  k_gemm<<<dim3(16, 64), 256, 0, stream>>>(xc, wb, rows, bias, out);
}
